// MultiHeadAttention_67826123539068
// MI455X (gfx1250) — matrix operand order repaired and checked by reading, not compiled, not run on hardware
//
#include <hip/hip_runtime.h>
#include <hip/hip_bf16.h>

typedef __attribute__((ext_vector_type(16))) _Float16 v16h;
typedef __attribute__((ext_vector_type(8)))  _Float16 v8h;
typedef __attribute__((ext_vector_type(8)))  float    v8f;

#define MODEL_DIM 512
#define HEAD_NUM  8
#define HEAD_DIM  64
#define BATCH     4
#define SEQ       2048
#define NROW      (BATCH * SEQ)   // 8192

// ---------------------------------------------------------------------------
// WMMA wrapper: D = A(16x32 f16) x B(32x16 f16) + C(16x16 f32)
// ---------------------------------------------------------------------------
__device__ __forceinline__ v8f wmma_f16f32(v16h a, v16h b, v8f c) {
  return __builtin_amdgcn_wmma_f32_16x16x32_f16(
      /*neg_a=*/false, a, /*neg_b=*/false, b,
      /*c_mod=*/(short)0, c, /*reuse_a=*/false, /*reuse_b=*/false);
}

// ---------------------------------------------------------------------------
// CDNA5 async copy: global -> LDS, 16B per lane, tracked by ASYNCcnt.
// Low 32 bits of a generic pointer to __shared__ are the LDS byte address.
// ---------------------------------------------------------------------------
__device__ __forceinline__ void async_load_b128(void* lds_ptr, const void* gptr) {
  unsigned lds_off = (unsigned)(uintptr_t)lds_ptr;
  asm volatile("global_load_async_to_lds_b128 %0, %1, off"
               :: "v"(lds_off), "v"(gptr) : "memory");
}
__device__ __forceinline__ void wait_asynccnt0() {
  asm volatile("s_wait_asynccnt 0" ::: "memory");
}

// ---------------------------------------------------------------------------
// Fragment loaders. All lane logic done by the caller; these just gather.
// A-frag (16x32 f16): halves 0..7 -> k=klo8+0..7, halves 8..15 -> k=klo8+16..23
// B-frag (32x16 f16): same order as A, k=klo8+0..7 then klo8+16..23 (frag_8_8)
// ---------------------------------------------------------------------------
__device__ __forceinline__ v16h frag_8_8(const _Float16* p) {
  v16h a;
#pragma unroll
  for (int t = 0; t < 8; ++t) a[t] = p[t];
#pragma unroll
  for (int t = 0; t < 8; ++t) a[8 + t] = p[16 + t];
  return a;
}

__device__ __forceinline__ v16h frag_8_8_f32(const float* p) {
  v16h a;
#pragma unroll
  for (int t = 0; t < 8; ++t) a[t] = (_Float16)p[t];
#pragma unroll
  for (int t = 0; t < 8; ++t) a[8 + t] = (_Float16)p[16 + t];
  return a;
}

__device__ __forceinline__ v16h frag_16(const _Float16* p) {
  v16h a;
#pragma unroll
  for (int t = 0; t < 16; ++t) a[t] = p[t];
  return a;
}

// ---------------------------------------------------------------------------
// fp32 -> f16 elementwise conversion (weights)
// ---------------------------------------------------------------------------
__global__ void cvt_kernel(const float* __restrict__ src,
                           _Float16* __restrict__ dst, int n) {
  int i = blockIdx.x * blockDim.x + threadIdx.x;
  if (i < n) dst[i] = (_Float16)src[i];
}

// ---------------------------------------------------------------------------
// Projection GEMM: Out[M,512](f16) = scale * (X[M,512](f32) @ W[512,512]^T)
// Wave register tile = 16 rows x 64 cols; A-fragment reused across 4 WMMAs
// per k-step. All four B-fragments are loaded up front so loadcnt waits can
// be staged under the matrix ops. Block = 4 waves = 64 rows x 64 cols.
// ---------------------------------------------------------------------------
__global__ void __launch_bounds__(128)
proj_f32_kernel(const float* __restrict__ X, const _Float16* __restrict__ W,
                _Float16* __restrict__ Out, float scale) {
  const int lane  = threadIdx.x & 31;
  const int wave  = threadIdx.x >> 5;
  const int row0  = (blockIdx.x * 4 + wave) * 16;
  const int col0  = blockIdx.y * 64;
  const int nl    = lane & 15;
  const int klo8  = (lane < 16) ? 0 : 8;
  const int klo16 = (lane < 16) ? 0 : 16;

  const float*    arow = X + (size_t)(row0 + nl) * MODEL_DIM + klo8;
  const _Float16* wrow = W + (size_t)(col0 + nl) * MODEL_DIM + klo8;

  v8f acc[4] = {};
#pragma unroll 2
  for (int kk = 0; kk < MODEL_DIM; kk += 32) {
    const v16h a  = frag_8_8_f32(arow + kk);
    const v16h b0 = frag_8_8(wrow + (size_t)0 * 16 * MODEL_DIM + kk);
    const v16h b1 = frag_8_8(wrow + (size_t)1 * 16 * MODEL_DIM + kk);
    const v16h b2 = frag_8_8(wrow + (size_t)2 * 16 * MODEL_DIM + kk);
    const v16h b3 = frag_8_8(wrow + (size_t)3 * 16 * MODEL_DIM + kk);
    acc[0] = wmma_f16f32(a, b0, acc[0]);
    acc[1] = wmma_f16f32(a, b1, acc[1]);
    acc[2] = wmma_f16f32(a, b2, acc[2]);
    acc[3] = wmma_f16f32(a, b3, acc[3]);
  }

  const int mb = (lane < 16) ? 0 : 8;
#pragma unroll
  for (int c = 0; c < 4; ++c)
#pragma unroll
    for (int r = 0; r < 8; ++r)
      Out[(size_t)(row0 + mb + r) * MODEL_DIM + col0 + c * 16 + nl] =
          (_Float16)(acc[c][r] * scale);
}

// ---------------------------------------------------------------------------
// Output projection: Out[M,512](f32) = A[M,512](f16) @ Wo[512,512]^T
// ---------------------------------------------------------------------------
__global__ void __launch_bounds__(128)
out_proj_kernel(const _Float16* __restrict__ A, const _Float16* __restrict__ W,
                float* __restrict__ Out) {
  const int lane  = threadIdx.x & 31;
  const int wave  = threadIdx.x >> 5;
  const int row0  = (blockIdx.x * 4 + wave) * 16;
  const int col0  = blockIdx.y * 64;
  const int nl    = lane & 15;
  const int klo8  = (lane < 16) ? 0 : 8;
  const int klo16 = (lane < 16) ? 0 : 16;

  const _Float16* arow = A + (size_t)(row0 + nl) * MODEL_DIM + klo8;
  const _Float16* wrow = W + (size_t)(col0 + nl) * MODEL_DIM + klo8;

  v8f acc[4] = {};
#pragma unroll 2
  for (int kk = 0; kk < MODEL_DIM; kk += 32) {
    const v16h a  = frag_8_8(arow + kk);
    const v16h b0 = frag_8_8(wrow + (size_t)0 * 16 * MODEL_DIM + kk);
    const v16h b1 = frag_8_8(wrow + (size_t)1 * 16 * MODEL_DIM + kk);
    const v16h b2 = frag_8_8(wrow + (size_t)2 * 16 * MODEL_DIM + kk);
    const v16h b3 = frag_8_8(wrow + (size_t)3 * 16 * MODEL_DIM + kk);
    acc[0] = wmma_f16f32(a, b0, acc[0]);
    acc[1] = wmma_f16f32(a, b1, acc[1]);
    acc[2] = wmma_f16f32(a, b2, acc[2]);
    acc[3] = wmma_f16f32(a, b3, acc[3]);
  }

  const int mb = (lane < 16) ? 0 : 8;
#pragma unroll
  for (int c = 0; c < 4; ++c)
#pragma unroll
    for (int r = 0; r < 8; ++r)
      Out[(size_t)(row0 + mb + r) * MODEL_DIM + col0 + c * 16 + nl] = acc[c][r];
}

// ---------------------------------------------------------------------------
// Stage one 64-key chunk: K rows async-DMA'd to LDS (ASYNCcnt), V rows
// loaded 16B/lane and scattered transposed into LDS.
// ---------------------------------------------------------------------------
__device__ __forceinline__ void stage_chunk(_Float16 (*Kt)[64], _Float16 (*Vt)[72],
                                            const _Float16* __restrict__ Kp,
                                            const _Float16* __restrict__ Vp,
                                            size_t base, int s0, int tid) {
#pragma unroll
  for (int j = 0; j < 4; ++j) {
    const int idx = tid + j * 128;             // 0..511
    const int r = idx >> 3;                    // 0..63
    const int c = (idx & 7) * 8;               // 0..56
    const size_t g = base + (size_t)(s0 + r) * MODEL_DIM + c;
    async_load_b128(&Kt[r][c], Kp + g);
    const v8h vv = *(const v8h*)(Vp + g);
#pragma unroll
    for (int t = 0; t < 8; ++t) Vt[c + t][r] = vv[t];
  }
}

// ---------------------------------------------------------------------------
// Flash attention over one (b, h): block = 4 waves = 64 query rows,
// 64-key chunks, double-buffered LDS staging so the async DMA of chunk n+1
// overlaps the 16 WMMAs of chunk n; one barrier per chunk.
// Wave computes S^T = K_chunk @ Q^T so softmax row-state is lane-indexed and
// exp(S^T) repacks into the two P A-fragments with zero shuffles.
// ---------------------------------------------------------------------------
__global__ void __launch_bounds__(128)
flash_attn_kernel(const _Float16* __restrict__ Qp, const _Float16* __restrict__ Kp,
                  const _Float16* __restrict__ Vp, _Float16* __restrict__ Op) {
  __shared__ __align__(16) _Float16 KtBuf[2][64][64];   // 2 x 8 KB
  __shared__ __align__(16) _Float16 VtBuf[2][64][72];   // 2 x 9 KB (transposed+pad)

  const int lane  = threadIdx.x & 31;
  const int wave  = threadIdx.x >> 5;
  const int bb    = blockIdx.z;
  const int hh    = blockIdx.y;
  const int q0    = blockIdx.x * 64 + wave * 16;
  const size_t base = (size_t)bb * SEQ * MODEL_DIM + (size_t)hh * HEAD_DIM;

  const int nl    = lane & 15;
  const int klo8  = (lane < 16) ? 0 : 8;
  const int klo16 = (lane < 16) ? 0 : 16;
  const int mb    = (lane < 16) ? 0 : 8;

  // Q^T B-fragments (fixed for whole kernel): B(k=d, n=query row)
  const _Float16* qrow = Qp + base + (size_t)(q0 + nl) * MODEL_DIM;
  const v16h bq0 = frag_8_8(qrow + klo8);        // d = 0..31
  const v16h bq1 = frag_8_8(qrow + 32 + klo8);   // d = 32..63

  v8f acc[4] = {};            // out tiles (m x d), d-tiles 0..3
  float run_max = -1e30f;     // per lane: query row = lane & 15
  float run_sum = 0.0f;       // per half-lane partial (32 of 64 keys/chunk)

  // per-thread prefetch coordinates (one cacheline region per thread)
  const int pr = threadIdx.x >> 1;          // 0..63
  const int pc = (threadIdx.x & 1) * 32;

  // prologue: stage chunk 0 into buffer 0
  stage_chunk(KtBuf[0], VtBuf[0], Kp, Vp, base, 0, threadIdx.x);
  wait_asynccnt0();
  __syncthreads();

  int cur = 0;
  for (int s0 = 0; s0 < SEQ; s0 += 64) {
    const int nxt = cur ^ 1;
    // overlap: DMA chunk n+1 into the other buffer while computing chunk n
    if (s0 + 64 < SEQ)
      stage_chunk(KtBuf[nxt], VtBuf[nxt], Kp, Vp, base, s0 + 64, threadIdx.x);
    // software prefetch two chunks ahead into cache
    if (s0 + 128 < SEQ) {
      const size_t gn = base + (size_t)(s0 + 128 + pr) * MODEL_DIM + pc;
      __builtin_prefetch(Kp + gn, 0, 3);
      __builtin_prefetch(Vp + gn, 0, 3);
    }

    _Float16 (*Kt)[64] = KtBuf[cur];
    _Float16 (*Vt)[72] = VtBuf[cur];

    // scores^T = K_chunk @ Q^T : 4 key tiles x (2 WMMA over d)
    v8f scf[4];
#pragma unroll
    for (int kt = 0; kt < 4; ++kt) {
      v8f s = {};
      s = wmma_f16f32(frag_8_8(&Kt[kt * 16 + nl][klo8]),      bq0, s);
      s = wmma_f16f32(frag_8_8(&Kt[kt * 16 + nl][32 + klo8]), bq1, s);
      scf[kt] = s;
    }

    // Online softmax: chunk max (combine half-lanes with one xor16)
    float cm = -1e30f;
#pragma unroll
    for (int kt = 0; kt < 4; ++kt)
#pragma unroll
      for (int r = 0; r < 8; ++r) cm = fmaxf(cm, scf[kt][r]);
    cm = fmaxf(cm, __shfl_xor(cm, 16, 32));
    const float nm   = fmaxf(run_max, cm);
    const float resc = __expf(run_max - nm);
    run_max = nm;
    run_sum *= resc;

    // exp + pack directly into two P A-fragments (keys 0..31 and 32..63)
    v16h ap0, ap1;
    float ls = 0.0f;
#pragma unroll
    for (int r = 0; r < 8; ++r) {
      const float p0 = __expf(scf[0][r] - nm);
      const float p1 = __expf(scf[1][r] - nm);
      const float p2 = __expf(scf[2][r] - nm);
      const float p3 = __expf(scf[3][r] - nm);
      ls += (p0 + p1) + (p2 + p3);
      ap0[r] = (_Float16)p0; ap0[8 + r] = (_Float16)p1;
      ap1[r] = (_Float16)p2; ap1[8 + r] = (_Float16)p3;
    }
    run_sum += ls;

    // Rescale output accumulators (rows are VGPR-indexed in the out frag)
#pragma unroll
    for (int r = 0; r < 8; ++r) {
      const float srw = __shfl(resc, mb + r, 32);
      acc[0][r] *= srw; acc[1][r] *= srw; acc[2][r] *= srw; acc[3][r] *= srw;
    }

    // out += P(16x64) @ V_chunk(64x64): B(k,d) = Vt[d][k], contiguous in k
#pragma unroll
    for (int dt = 0; dt < 4; ++dt) {
      acc[dt] = wmma_f16f32(ap0, frag_8_8(&Vt[dt * 16 + nl][klo8]),      acc[dt]);
      acc[dt] = wmma_f16f32(ap1, frag_8_8(&Vt[dt * 16 + nl][32 + klo8]), acc[dt]);
    }

    // publish next buffer (async K done + V stores visible to all waves),
    // and confirm everyone is finished reading buf[cur] before its reuse
    wait_asynccnt0();
    __syncthreads();
    cur = nxt;
  }

  // Combine half-lane partial sums, normalize, store (f16 into head slice)
  run_sum += __shfl_xor(run_sum, 16, 32);
#pragma unroll
  for (int r = 0; r < 8; ++r) {
    const float inv = 1.0f / __shfl(run_sum, mb + r, 32);
    _Float16* orow = Op + base + (size_t)(q0 + mb + r) * MODEL_DIM;
    orow[nl]      = (_Float16)(acc[0][r] * inv);
    orow[16 + nl] = (_Float16)(acc[1][r] * inv);
    orow[32 + nl] = (_Float16)(acc[2][r] * inv);
    orow[48 + nl] = (_Float16)(acc[3][r] * inv);
  }
}

// ---------------------------------------------------------------------------
// Launcher
// ---------------------------------------------------------------------------
extern "C" void kernel_launch(void* const* d_in, const int* in_sizes, int n_in,
                              void* d_out, int out_size, void* d_ws, size_t ws_size,
                              hipStream_t stream) {
  (void)in_sizes; (void)n_in; (void)out_size; (void)ws_size;

  const float* q  = (const float*)d_in[0];
  const float* kx = (const float*)d_in[1];
  const float* vx = (const float*)d_in[2];
  const float* Wq = (const float*)d_in[3];
  const float* Wk = (const float*)d_in[4];
  const float* Wv = (const float*)d_in[5];
  const float* Wo = (const float*)d_in[6];
  float* out = (float*)d_out;

  // Workspace carve-up (all f16): 4 weights + Qp/Kp/Vp/attended ≈ 36 MB
  _Float16* p = (_Float16*)d_ws;
  const size_t WSZ = (size_t)MODEL_DIM * MODEL_DIM;  // 262144
  const size_t ASZ = (size_t)NROW * MODEL_DIM;       // 4194304
  _Float16* Wq16 = p; p += WSZ;
  _Float16* Wk16 = p; p += WSZ;
  _Float16* Wv16 = p; p += WSZ;
  _Float16* Wo16 = p; p += WSZ;
  _Float16* Qp   = p; p += ASZ;
  _Float16* Kp   = p; p += ASZ;
  _Float16* Vp   = p; p += ASZ;
  _Float16* Ap   = p;

  // 1) weights fp32 -> f16
  dim3 cb(256), cg((unsigned)(WSZ / 256));
  cvt_kernel<<<cg, cb, 0, stream>>>(Wq, Wq16, (int)WSZ);
  cvt_kernel<<<cg, cb, 0, stream>>>(Wk, Wk16, (int)WSZ);
  cvt_kernel<<<cg, cb, 0, stream>>>(Wv, Wv16, (int)WSZ);
  cvt_kernel<<<cg, cb, 0, stream>>>(Wo, Wo16, (int)WSZ);

  // 2) projections (Q pre-scaled by 1/sqrt(HEAD_DIM) = 0.125)
  dim3 gb(128), gg(NROW / 64, MODEL_DIM / 64);
  proj_f32_kernel<<<gg, gb, 0, stream>>>(q,  Wq16, Qp, 0.125f);
  proj_f32_kernel<<<gg, gb, 0, stream>>>(kx, Wk16, Kp, 1.0f);
  proj_f32_kernel<<<gg, gb, 0, stream>>>(vx, Wv16, Vp, 1.0f);

  // 3) flash attention per (b, h): 64 query rows/block, 64-key chunks,
  //    double-buffered async LDS staging
  flash_attn_kernel<<<dim3(SEQ / 64, HEAD_NUM, BATCH), gb, 0, stream>>>(
      Qp, Kp, Vp, Ap);

  // 4) output projection -> fp32
  out_proj_kernel<<<gg, gb, 0, stream>>>(Ap, Wo16, out);
}
